// prop_timestamp_22728966930663
// MI455X (gfx1250) — matrix operand order repaired and checked by reading, not compiled, not run on hardware
//
#include <hip/hip_runtime.h>

// ---------------------------------------------------------------------------
// Problem constants (from reference): x (256,32,207,12) f32, adj (207,207),
// W (32,32), b (32). GDEP=3, ALPHA=0.05.
// Diffusion collapsed to single GEMM with P = c0*I + c1*A + c2*A^2 + c3*A^3.
// ---------------------------------------------------------------------------
#define BB   256
#define CC   32
#define NN   207
#define LL   12
#define NP   224          // N padded: 14 M-tiles of 16 / 7 K-chunks of 32
#define BLS  (BB*LL)      // 3072 (b,l) slices for GEMM1
#define VTP  7            // v-tile PAIRS (14 tiles of 16 = 224 rows, rows>=207 zero)
#define BLP  (BLS/2)      // 1536 (b,l) slice pairs
#define ROWS (NN*LL)      // 2484 (v,l) rows per batch for GEMM2
#define ROWST 2496        // padded row stride (multiple of 16)
#define RT   156          // ceil(2484/16) row tiles

typedef __attribute__((ext_vector_type(8)))  float   v8f;
typedef __attribute__((ext_vector_type(8)))  __bf16  v8bf;
typedef __attribute__((ext_vector_type(16))) __bf16  v16bf;

static __device__ inline unsigned short f2bf(float f) {
  unsigned int u = __float_as_uint(f);
  unsigned int r = (u + 0x7FFFu + ((u >> 16) & 1u)) >> 16;   // RNE
  return (unsigned short)r;
}

static __device__ inline v16bf cat16(v8bf lo, v8bf hi) {
  return __builtin_shufflevector(lo, hi, 0,1,2,3,4,5,6,7,8,9,10,11,12,13,14,15);
}

// ---------------------------------------------------------------------------
// 1) Row-normalized adjacency with self loops: Af[v][w] = (adj+I)/rowsum
// ---------------------------------------------------------------------------
__global__ __launch_bounds__(256) void normalize_adj(const float* __restrict__ adj,
                                                     float* __restrict__ Af) {
  __shared__ float sbuf[256];
  int v = blockIdx.x;          // 0..206
  int i = threadIdx.x;
  float val = 0.f;
  if (i < NN) val = adj[v * NN + i] + (i == v ? 1.f : 0.f);
  sbuf[i] = val;
  __syncthreads();
  for (int s = 128; s > 0; s >>= 1) {
    if (i < s) sbuf[i] += sbuf[i + s];
    __syncthreads();
  }
  float inv = 1.f / sbuf[0];
  if (i < NN) Af[v * NN + i] = val * inv;
}

// ---------------------------------------------------------------------------
// 2) Tiny fp32 GEMM 207x207x207 (used twice: A^2, A^3). ~8.9 MFLOP each.
// ---------------------------------------------------------------------------
__global__ __launch_bounds__(256) void mm207(const float* __restrict__ A1,
                                             const float* __restrict__ A2,
                                             float* __restrict__ Cm) {
  int idx = blockIdx.x * 256 + threadIdx.x;
  if (idx >= NN * NN) return;
  int v = idx / NN, w = idx % NN;
  float s = 0.f;
  for (int k = 0; k < NN; ++k) s += A1[v * NN + k] * A2[k * NN + w];
  Cm[idx] = s;
}

// ---------------------------------------------------------------------------
// 3) P = 0.05 I + 0.0475 A + 0.045125 A^2 + 0.857375 A^3  -> bf16, 224x224 padded
// ---------------------------------------------------------------------------
__global__ __launch_bounds__(256) void build_P(const float* __restrict__ Af,
                                               const float* __restrict__ T1,
                                               const float* __restrict__ T2,
                                               unsigned short* __restrict__ Pb) {
  int idx = blockIdx.x * 256 + threadIdx.x;   // 0..224*224-1
  if (idx >= NP * NP) return;
  int v = idx / NP, k = idx % NP;
  float p = 0.f;
  if (v < NN && k < NN) {
    int j = v * NN + k;
    p = 0.857375f * T2[j] + 0.045125f * T1[j] + 0.0475f * Af[j]
        + (v == k ? 0.05f : 0.f);
  }
  Pb[idx] = f2bf(p);
}

__global__ __launch_bounds__(256) void conv_W(const float* __restrict__ W,
                                              unsigned short* __restrict__ Wb) {
  int idx = blockIdx.x * 256 + threadIdx.x;
  if (idx < CC * CC) Wb[idx] = f2bf(W[idx]);   // row-major [o][c]
}

// ---------------------------------------------------------------------------
// 4) Transpose x (B,C,N,L) f32 -> Xt[b][l][c][w(224 pad)] bf16 (K-contiguous B op)
// ---------------------------------------------------------------------------
__global__ __launch_bounds__(256) void xpose(const float* __restrict__ x,
                                             unsigned short* __restrict__ Xt) {
  long long idx = (long long)blockIdx.x * 256 + threadIdx.x;  // 22,020,096 total
  int w = (int)(idx % NP);
  int c = (int)((idx / NP) % CC);
  int l = (int)((idx / (NP * CC)) % LL);
  int b = (int)(idx / ((long long)NP * CC * LL));
  float val = 0.f;
  if (w < NN) val = x[(((long long)b * CC + c) * NN + w) * LL + l];
  Xt[idx] = f2bf(val);
}

// ---------------------------------------------------------------------------
// 5) GEMM1 (WMMA bf16): per (b,l):  H3[v, c] = P(207x207) * Xl(207x32)
//    Register-blocked: one wave = 2 v-tiles x 2 (b,l) slices x 2 c-tiles
//    = 8 f32 accumulators, A fragments reused 4x, B fragments reused 2x.
//    K-loop: 7 x (16x16x32 bf16) -> 56 WMMAs per wave.
//    Output: H3t[(b*2496 + v*12 + l)][c]  (bf16, A-operand-friendly for GEMM2)
// ---------------------------------------------------------------------------
__global__ __launch_bounds__(256) void gemm1(const unsigned short* __restrict__ Pb,
                                             const unsigned short* __restrict__ Xt,
                                             unsigned short* __restrict__ H3t) {
  int wid  = blockIdx.x * 8 + (threadIdx.x >> 5);   // 0..10751 == 7*1536
  int lane = threadIdx.x & 31;
  int vtp = wid % VTP;            // v-tile pair: tiles 2*vtp, 2*vtp+1
  int blp = wid / VTP;            // (b,l) pair: slices 2*blp, 2*blp+1 (same b)
  int half = lane >> 4;
  int ln   = lane & 15;

  const __bf16* P = (const __bf16*)Pb;
  const __bf16* X = (const __bf16*)Xt;

  int vt0 = vtp * 2;
  int bl0 = blp * 2;
  const __bf16* ap0  = P + (vt0 * 16 + ln) * NP + half * 16;  // A frag, v-tile 0
  const __bf16* ap1  = ap0 + 16 * NP;                          // A frag, v-tile 1
  const __bf16* bp00 = X + ((long long)bl0 * CC + ln)      * NP + half * 16; // bl0,c-tile0
  const __bf16* bp01 = bp00 + 16 * NP;                                       // bl0,c-tile1
  const __bf16* bp10 = bp00 + CC * NP;                                       // bl1,c-tile0
  const __bf16* bp11 = bp10 + 16 * NP;                                       // bl1,c-tile1

  v8f acc000 = {}, acc001 = {}, acc010 = {}, acc011 = {};  // acc[vt][bl][ct]
  v8f acc100 = {}, acc101 = {}, acc110 = {}, acc111 = {};
#pragma unroll
  for (int kk = 0; kk < 7; ++kk) {
    int ko = kk * 32;
    v16bf a0  = cat16(*(const v8bf*)(ap0  + ko), *(const v8bf*)(ap0  + ko + 8));
    v16bf a1  = cat16(*(const v8bf*)(ap1  + ko), *(const v8bf*)(ap1  + ko + 8));
    v16bf b00 = cat16(*(const v8bf*)(bp00 + ko), *(const v8bf*)(bp00 + ko + 8));
    v16bf b01 = cat16(*(const v8bf*)(bp01 + ko), *(const v8bf*)(bp01 + ko + 8));
    v16bf b10 = cat16(*(const v8bf*)(bp10 + ko), *(const v8bf*)(bp10 + ko + 8));
    v16bf b11 = cat16(*(const v8bf*)(bp11 + ko), *(const v8bf*)(bp11 + ko + 8));
    acc000 = __builtin_amdgcn_wmma_f32_16x16x32_bf16(false, a0, false, b00, (short)0, acc000, false, false);
    acc001 = __builtin_amdgcn_wmma_f32_16x16x32_bf16(false, a0, false, b01, (short)0, acc001, false, false);
    acc010 = __builtin_amdgcn_wmma_f32_16x16x32_bf16(false, a0, false, b10, (short)0, acc010, false, false);
    acc011 = __builtin_amdgcn_wmma_f32_16x16x32_bf16(false, a0, false, b11, (short)0, acc011, false, false);
    acc100 = __builtin_amdgcn_wmma_f32_16x16x32_bf16(false, a1, false, b00, (short)0, acc100, false, false);
    acc101 = __builtin_amdgcn_wmma_f32_16x16x32_bf16(false, a1, false, b01, (short)0, acc101, false, false);
    acc110 = __builtin_amdgcn_wmma_f32_16x16x32_bf16(false, a1, false, b10, (short)0, acc110, false, false);
    acc111 = __builtin_amdgcn_wmma_f32_16x16x32_bf16(false, a1, false, b11, (short)0, acc111, false, false);
  }

  int b  = bl0 / LL;              // bl0, bl0+1 share the batch (12 l's, even)
  int l0 = bl0 % LL;
  // D layout: lanes<16 hold rows r, lanes>=16 hold rows r+8; col = ln (+16 for ct1)
#pragma unroll
  for (int vt = 0; vt < 2; ++vt) {
    int vbase = (vt0 + vt) * 16 + half * 8;
    const v8f* a_bl0_c0 = vt ? &acc100 : &acc000;
    const v8f* a_bl0_c1 = vt ? &acc101 : &acc001;
    const v8f* a_bl1_c0 = vt ? &acc110 : &acc010;
    const v8f* a_bl1_c1 = vt ? &acc111 : &acc011;
#pragma unroll
    for (int r = 0; r < 8; ++r) {
      int v = vbase + r;
      if (v < NN) {
        long long row0 = ((long long)b * ROWST + v * LL + l0) * CC;       // slice bl0
        long long row1 = row0 + CC;                                       // slice bl0+1
        H3t[row0 + ln]      = f2bf((*a_bl0_c0)[r]);
        H3t[row0 + 16 + ln] = f2bf((*a_bl0_c1)[r]);
        H3t[row1 + ln]      = f2bf((*a_bl1_c0)[r]);
        H3t[row1 + 16 + ln] = f2bf((*a_bl1_c1)[r]);
      }
    }
  }
}

// ---------------------------------------------------------------------------
// 6) GEMM2 (WMMA bf16): out[(v,l), o] = H3t[(v,l), c] * W[o, c] + bias[o]
//    K = 32 -> single chunk, 2 WMMAs per wave. fp32 result straight to d_out.
// ---------------------------------------------------------------------------
__global__ __launch_bounds__(256) void gemm2(const unsigned short* __restrict__ H3t,
                                             const unsigned short* __restrict__ Wb,
                                             const float* __restrict__ bias,
                                             float* __restrict__ out) {
  int wid  = blockIdx.x * 8 + (threadIdx.x >> 5);  // 0..39935 == 156*256
  int lane = threadIdx.x & 31;
  int rt = wid % RT;
  int b  = wid / RT;
  int half = lane >> 4;
  int ln   = lane & 15;
  int row0 = rt * 16;

  int arow = row0 + ln;
  if (arow > ROWS - 1) arow = ROWS - 1;            // clamp: padded rows never read

  const __bf16* H  = (const __bf16*)H3t;
  const __bf16* Wp = (const __bf16*)Wb;

  const __bf16* ap  = H + ((long long)b * ROWST + arow) * CC + half * 16;
  v16bf a = cat16(*(const v8bf*)(ap), *(const v8bf*)(ap + 8));

  const __bf16* wp0 = Wp + ln * CC + half * 16;           // col o = ln
  const __bf16* wp1 = Wp + (16 + ln) * CC + half * 16;    // col o = 16+ln
  v16bf b0 = cat16(*(const v8bf*)(wp0), *(const v8bf*)(wp0 + 8));
  v16bf b1 = cat16(*(const v8bf*)(wp1), *(const v8bf*)(wp1 + 8));

  v8f acc0 = {}; v8f acc1 = {};
  acc0 = __builtin_amdgcn_wmma_f32_16x16x32_bf16(false, a, false, b0,
                                                 (short)0, acc0, false, false);
  acc1 = __builtin_amdgcn_wmma_f32_16x16x32_bf16(false, a, false, b1,
                                                 (short)0, acc1, false, false);

  float bv0 = bias[ln];
  float bv1 = bias[16 + ln];
  int m0 = row0 + half * 8;
  float* o0 = out + ((long long)b * CC + ln)      * ROWS;
  float* o1 = out + ((long long)b * CC + 16 + ln) * ROWS;
#pragma unroll
  for (int r = 0; r < 8; ++r) {
    int m = m0 + r;
    if (m < ROWS) {
      o0[m] = acc0[r] + bv0;
      o1[m] = acc1[r] + bv1;
    }
  }
}

// ---------------------------------------------------------------------------
// Launch
// ---------------------------------------------------------------------------
extern "C" void kernel_launch(void* const* d_in, const int* in_sizes, int n_in,
                              void* d_out, int out_size, void* d_ws, size_t ws_size,
                              hipStream_t stream) {
  const float* x    = (const float*)d_in[0];
  const float* adj  = (const float*)d_in[1];
  const float* W    = (const float*)d_in[2];
  const float* bias = (const float*)d_in[3];
  float* out = (float*)d_out;

  char* ws = (char*)d_ws;
  // Workspace layout (256B aligned offsets), ~85.6 MB total
  float*          Af  = (float*)(ws + 0);                 // 207*207*4 = 171396
  float*          T1  = (float*)(ws + 171520);            // A^2
  float*          T2  = (float*)(ws + 343040);            // A^3
  unsigned short* Pb  = (unsigned short*)(ws + 514560);   // 224*224*2 = 100352
  unsigned short* Wb  = (unsigned short*)(ws + 614912);   // 32*32*2   = 2048
  unsigned short* Xt  = (unsigned short*)(ws + 616960);   // 256*12*32*224*2 = 44040192
  unsigned short* H3t = (unsigned short*)(ws + 44657152); // 256*2496*32*2   = 40894464

  normalize_adj<<<NN, 256, 0, stream>>>(adj, Af);
  mm207<<<(NN * NN + 255) / 256, 256, 0, stream>>>(Af, Af, T1);      // A^2
  mm207<<<(NN * NN + 255) / 256, 256, 0, stream>>>(Af, T1, T2);      // A^3
  build_P<<<(NP * NP) / 256, 256, 0, stream>>>(Af, T1, T2, Pb);      // 196 blocks
  conv_W<<<4, 256, 0, stream>>>(W, Wb);
  xpose<<<(BB * LL * CC * NP) / 256, 256, 0, stream>>>(x, Xt);       // 86016 blocks
  gemm1<<<(VTP * BLP) / 8, 256, 0, stream>>>(Pb, Xt, H3t);           // 1344 blocks
  gemm2<<<(RT * BB) / 8, 256, 0, stream>>>(H3t, Wb, bias, out);      // 4992 blocks
}
